// AttEncoderLayer_86174223827266
// MI455X (gfx1250) — matrix operand order repaired and checked by reading, not compiled, not run on hardware
//
#include <hip/hip_runtime.h>
#include <cstdint>

// ---------------------------------------------------------------------------
// MI455X (gfx1250) transformer encoder layer.
// All GEMMs via v_wmma_f32_16x16x32_bf16 (bf16 in, f32 accumulate).
// ---------------------------------------------------------------------------

#define B_   8
#define L_   1024
#define D_   256
#define H_   8
#define DK_  32
#define DV_  32
#define DF_  1024
#define M_   (B_ * L_)        // 8192 token rows

typedef __attribute__((ext_vector_type(16))) __bf16 v16bf;
typedef __attribute__((ext_vector_type(8)))  float  v8f;

union FragU { v16bf v; uint4 q[2]; };

__device__ __forceinline__ uint16_t f2bf(float f) {
    uint32_t u = __float_as_uint(f);
    uint32_t r = u + 0x7FFFu + ((u >> 16) & 1u);   // round-to-nearest-even
    return (uint16_t)(r >> 16);
}

// A fragment: 16x32 bf16, M x K, row-major A. lane -> row (l&15),
// K chunks at kofs+(l>>4)*8 and +16 (per ISA 16-bit A layout).
__device__ __forceinline__ v16bf load_a_frag(const uint16_t* A, int lda,
                                             int m0, int kofs, int lane) {
    int m  = m0 + (lane & 15);
    int ko = kofs + ((lane >> 4) << 3);
    const uint16_t* p = A + (size_t)m * lda + ko;
    FragU f;
    f.q[0] = *reinterpret_cast<const uint4*>(p);
    f.q[1] = *reinterpret_cast<const uint4*>(p + 16);
    return f.v;
}

// B fragment: 32x16 bf16 (K x N) fed from BT stored N x K row-major.
// lane -> col (l&15), K chunks at kofs+(l>>4)*8 and +16 (same as A).
__device__ __forceinline__ v16bf load_bt_frag(const uint16_t* BT, int ldb,
                                              int n0, int kofs, int lane) {
    int n  = n0 + (lane & 15);
    int kb = kofs + ((lane >> 4) << 3);
    const uint4* p = reinterpret_cast<const uint4*>(BT + (size_t)n * ldb + kb);
    FragU f;
    f.q[0] = p[0];
    f.q[1] = p[2];
    return f.v;
}

#define WMMA_BF16(a, b, c) \
    __builtin_amdgcn_wmma_f32_16x16x32_bf16(false, (a), false, (b), (short)0, (c), false, false)

__device__ __forceinline__ float half_max16(float v) {
    v = fmaxf(v, __shfl_xor(v, 1, 32));
    v = fmaxf(v, __shfl_xor(v, 2, 32));
    v = fmaxf(v, __shfl_xor(v, 4, 32));
    v = fmaxf(v, __shfl_xor(v, 8, 32));
    return v;
}
__device__ __forceinline__ float half_sum16(float v) {
    v += __shfl_xor(v, 1, 32);
    v += __shfl_xor(v, 2, 32);
    v += __shfl_xor(v, 4, 32);
    v += __shfl_xor(v, 8, 32);
    return v;
}
__device__ __forceinline__ float full_sum32(float v) {
    v += __shfl_xor(v, 1, 32);
    v += __shfl_xor(v, 2, 32);
    v += __shfl_xor(v, 4, 32);
    v += __shfl_xor(v, 8, 32);
    v += __shfl_xor(v, 16, 32);
    return v;
}

// ------------------------- prep / conversion kernels ------------------------

__global__ void k_cvt_bf16(const float* __restrict__ s, uint16_t* __restrict__ d, int n) {
    int i = blockIdx.x * 256 + threadIdx.x;
    if (i < n) d[i] = f2bf(s[i]);
}

// (H, D, DK) f32 -> (H, DK, D) bf16   (N x K layout for the QKV GEMM)
__global__ void k_cvt_tr_qkv(const float* __restrict__ s, uint16_t* __restrict__ d, int n) {
    int i = blockIdx.x * 256 + threadIdx.x;
    if (i >= n) return;
    int h   = i >> 13;          // / (D_*DK_)
    int rem = i & 8191;
    int dd  = rem >> 5;         // / DK_
    int kk  = rem & 31;
    d[((size_t)h * DK_ + kk) * D_ + dd] = f2bf(s[i]);
}

// --------------------------------- QKV GEMM ---------------------------------
// wave -> (mat, h, mtile, ntile); C = X[16x256] * W_h^T, K-loop of 8 WMMAs.
// q,k stored [h][m][dk] bf16; v stored transposed [h][b][dv][l] bf16.
__global__ __launch_bounds__(128) void k_qkv(
    const uint16_t* __restrict__ xb,
    const uint16_t* __restrict__ wqt, const uint16_t* __restrict__ wkt,
    const uint16_t* __restrict__ wvt,
    uint16_t* __restrict__ qb, uint16_t* __restrict__ kb, uint16_t* __restrict__ vt) {
    int lane = threadIdx.x & 31;
    int wid  = blockIdx.x * 4 + (threadIdx.x >> 5);
    if (wid >= 3 * H_ * (M_ / 16) * 2) return;
    int nt  = wid & 1;
    int mt  = (wid >> 1) & 511;
    int h   = (wid >> 10) & 7;
    int mat = wid >> 13;

    const uint16_t* BT = (mat == 0 ? wqt : (mat == 1 ? wkt : wvt)) + (size_t)h * DK_ * D_;
    int m0 = mt * 16, n0 = nt * 16;

    v8f acc = {};
    for (int k = 0; k < D_; k += 32) {
        v16bf a = load_a_frag(xb, D_, m0, k, lane);
        v16bf b = load_bt_frag(BT, D_, n0, k, lane);
        acc = WMMA_BF16(a, b, acc);
    }
    int nl = lane & 15, hl = lane >> 4;
    for (int r = 0; r < 8; ++r) {
        int row = m0 + r + 8 * hl;
        int col = n0 + nl;
        uint16_t v = f2bf(acc[r]);
        if (mat == 0)      qb[((size_t)h * M_ + row) * DK_ + col] = v;
        else if (mat == 1) kb[((size_t)h * M_ + row) * DK_ + col] = v;
        else {
            int b_ = row >> 10, li = row & 1023;
            vt[(((size_t)(h * B_ + b_) * DV_) + col) * L_ + li] = v;
        }
    }
}

// ------------------------- flash attention per q-tile -----------------------
// One wave: 16 q rows x full DV=32, online softmax over 64 k-tiles of 16,
// processed 32 k-columns per iteration (2 score WMMAs + 2 PV WMMAs).
__global__ __launch_bounds__(128) void k_attn(
    const uint16_t* __restrict__ qb, const uint16_t* __restrict__ kb,
    const uint16_t* __restrict__ vt, const uint8_t* __restrict__ mask,
    uint16_t* __restrict__ attn_out) {
    __shared__ __align__(16) uint16_t ptile[4][16 * 32];
    int lane = threadIdx.x & 31;
    int w    = threadIdx.x >> 5;
    int wid  = blockIdx.x * 4 + w;          // 0..4095
    int qt = wid & 63, b = (wid >> 6) & 7, h = wid >> 9;

    const uint16_t* Q  = qb + (size_t)(h * B_ + b) * L_ * DK_;
    const uint16_t* K_ = kb + (size_t)(h * B_ + b) * L_ * DK_;
    const uint16_t* V  = vt + (size_t)(h * B_ + b) * DV_ * L_;
    int q0 = qt * 16;
    int nl = lane & 15, hl = lane >> 4;
    const uint8_t* mbase = mask + ((size_t)b * L_ + q0) * L_;

    v16bf qf = load_a_frag(Q, DK_, q0, 0, lane);   // dk=32: one K-step
    v8f accA = {}, accB = {};
    float mrow[8], lrow[8];
    #pragma unroll
    for (int r = 0; r < 8; ++r) { mrow[r] = -1.0e30f; lrow[r] = 0.0f; }

    for (int k0 = 0; k0 < L_; k0 += 32) {
        v16bf kfA = load_bt_frag(K_, DK_, 0, 0, lane) ;  // placeholder overwritten below
        kfA = load_bt_frag(K_ + (size_t)k0 * DK_, DK_, 0, 0, lane);
        v16bf kfB = load_bt_frag(K_ + (size_t)(k0 + 16) * DK_, DK_, 0, 0, lane);
        v8f z = {};
        v8f sA = WMMA_BF16(qf, kfA, z);
        v8f sB = WMMA_BF16(qf, kfB, z);

        #pragma unroll
        for (int r = 0; r < 8; ++r) {
            int rl = r + 8 * hl;
            float sa = sA[r] * 0.0625f;             // 1/sqrt(d_model)=1/16
            float sb = sB[r] * 0.0625f;
            if (mbase[(size_t)rl * L_ + k0 + nl])      sa = -1.0e30f;
            if (mbase[(size_t)rl * L_ + k0 + 16 + nl]) sb = -1.0e30f;
            float t  = half_max16(fmaxf(sa, sb));
            float mn = fmaxf(mrow[r], t);
            float alpha = __expf(mrow[r] - mn);
            float pa = (sa <= -1.0e29f) ? 0.0f : __expf(sa - mn);
            float pb = (sb <= -1.0e29f) ? 0.0f : __expf(sb - mn);
            lrow[r] = lrow[r] * alpha + half_sum16(pa + pb);
            mrow[r] = mn;
            accA[r] *= alpha;
            accB[r] *= alpha;
            ptile[w][rl * 32 + nl]      = f2bf(pa);
            ptile[w][rl * 32 + 16 + nl] = f2bf(pb);
        }
        // C-layout probs -> A-layout fragment via LDS (DS ops in-order per wave)
        v16bf pf  = load_a_frag(&ptile[w][0], 32, 0, 0, lane);
        v16bf vfA = load_bt_frag(V, L_, 0,  k0, lane);   // dv 0..15
        v16bf vfB = load_bt_frag(V, L_, 16, k0, lane);   // dv 16..31
        accA = WMMA_BF16(pf, vfA, accA);
        accB = WMMA_BF16(pf, vfB, accB);
    }

    #pragma unroll
    for (int r = 0; r < 8; ++r) {
        float inv = 1.0f / lrow[r];
        int row = q0 + r + 8 * hl;
        size_t base = ((size_t)b * L_ + row) * (H_ * DV_) + h * DV_;
        attn_out[base + nl]      = f2bf(accA[r] * inv);
        attn_out[base + 16 + nl] = f2bf(accB[r] * inv);
    }
}

// ------------------- generic 16x16-tile GEMM with epilogue ------------------
// C = A[MxK](bf16) * BT[NxK](bf16)^T (+bias)(+resid)(ReLU?) -> f32 and/or bf16
__global__ __launch_bounds__(128) void k_gemm_ep(
    const uint16_t* __restrict__ A, const uint16_t* __restrict__ BT,
    const float* __restrict__ bias, const float* __restrict__ resid,
    float* __restrict__ outF, uint16_t* __restrict__ outB,
    int Mi, int Ni, int Ki, int relu) {
    int lane = threadIdx.x & 31;
    int wid  = blockIdx.x * 4 + (threadIdx.x >> 5);
    int ntc = Ni >> 4;
    if (wid >= (Mi >> 4) * ntc) return;
    int mt = wid / ntc, nt = wid % ntc;
    int m0 = mt * 16, n0 = nt * 16;

    v8f acc = {};
    for (int k = 0; k < Ki; k += 32) {
        v16bf a = load_a_frag(A, Ki, m0, k, lane);
        v16bf b = load_bt_frag(BT, Ki, n0, k, lane);
        acc = WMMA_BF16(a, b, acc);
    }
    int nl = lane & 15, hl = lane >> 4;
    int col = n0 + nl;
    float bv = bias ? bias[col] : 0.0f;
    #pragma unroll
    for (int r = 0; r < 8; ++r) {
        int row = m0 + r + 8 * hl;
        float v = acc[r] + bv;
        if (resid) v += resid[(size_t)row * Ni + col];
        if (relu)  v  = fmaxf(v, 0.0f);
        if (outF)  outF[(size_t)row * Ni + col] = v;
        if (outB)  outB[(size_t)row * Ni + col] = f2bf(v);
    }
}

// ------------------- layer norm (torch std ddof=1, eps on sigma) ------------
__global__ __launch_bounds__(128) void k_ln(
    const float* __restrict__ Z, const float* __restrict__ g,
    const float* __restrict__ bt, float* __restrict__ outF,
    uint16_t* __restrict__ outB, int rows) {
    int lane = threadIdx.x & 31;
    int row  = blockIdx.x * 4 + (threadIdx.x >> 5);
    if (row >= rows) return;
    const float* z = Z + (size_t)row * D_;
    float v[8], s = 0.0f;
    #pragma unroll
    for (int i = 0; i < 8; ++i) { v[i] = z[lane + 32 * i]; s += v[i]; }
    float mu = full_sum32(s) * (1.0f / D_);
    float q = 0.0f;
    #pragma unroll
    for (int i = 0; i < 8; ++i) { float d = v[i] - mu; q += d * d; }
    float sigma = sqrtf(full_sum32(q) / (float)(D_ - 1));
    float inv = 1.0f / (sigma + 1.0e-3f);
    #pragma unroll
    for (int i = 0; i < 8; ++i) {
        int c = lane + 32 * i;
        float y = (v[i] - mu) * inv * g[c] + bt[c];
        outF[(size_t)row * D_ + c] = y;
        if (outB) outB[(size_t)row * D_ + c] = f2bf(y);
    }
}

// --------------------------------- launcher ---------------------------------

extern "C" void kernel_launch(void* const* d_in, const int* in_sizes, int n_in,
                              void* d_out, int out_size, void* d_ws, size_t ws_size,
                              hipStream_t stream) {
    (void)in_sizes; (void)n_in; (void)out_size; (void)ws_size;
    const float*   enc    = (const float*)d_in[0];
    const uint8_t* mask   = (const uint8_t*)d_in[1];
    const float*   w_qs   = (const float*)d_in[2];
    const float*   w_ks   = (const float*)d_in[3];
    const float*   w_vs   = (const float*)d_in[4];
    const float*   proj_w = (const float*)d_in[5];
    const float*   proj_b = (const float*)d_in[6];
    const float*   ln1_a  = (const float*)d_in[7];
    const float*   ln1_b  = (const float*)d_in[8];
    const float*   w1     = (const float*)d_in[9];
    const float*   b1     = (const float*)d_in[10];
    const float*   w2     = (const float*)d_in[11];
    const float*   b2     = (const float*)d_in[12];
    const float*   ln2_a  = (const float*)d_in[13];
    const float*   ln2_b  = (const float*)d_in[14];

    char* ws = (char*)d_ws;
    size_t off = 0;
    auto alloc = [&](size_t bytes) { size_t o = off; off = (off + bytes + 255) & ~(size_t)255; return o; };

    uint16_t* x_bf   = (uint16_t*)(ws + alloc((size_t)M_ * D_ * 2));
    uint16_t* wqt    = (uint16_t*)(ws + alloc((size_t)H_ * DK_ * D_ * 2));
    uint16_t* wkt    = (uint16_t*)(ws + alloc((size_t)H_ * DK_ * D_ * 2));
    uint16_t* wvt    = (uint16_t*)(ws + alloc((size_t)H_ * DV_ * D_ * 2));
    uint16_t* pwb    = (uint16_t*)(ws + alloc((size_t)D_ * H_ * DV_ * 2));
    uint16_t* w1b    = (uint16_t*)(ws + alloc((size_t)DF_ * D_ * 2));
    uint16_t* w2b    = (uint16_t*)(ws + alloc((size_t)D_ * DF_ * 2));
    uint16_t* q_bf   = (uint16_t*)(ws + alloc((size_t)H_ * M_ * DK_ * 2));
    uint16_t* k_bf   = (uint16_t*)(ws + alloc((size_t)H_ * M_ * DK_ * 2));
    uint16_t* v_t    = (uint16_t*)(ws + alloc((size_t)H_ * M_ * DV_ * 2));
    uint16_t* a_out  = (uint16_t*)(ws + alloc((size_t)M_ * D_ * 2));
    float*    z1     = (float*)   (ws + alloc((size_t)M_ * D_ * 4));
    float*    out1f  = (float*)   (ws + alloc((size_t)M_ * D_ * 4));
    uint16_t* out1b  = (uint16_t*)(ws + alloc((size_t)M_ * D_ * 2));
    uint16_t* hmid   = (uint16_t*)(ws + alloc((size_t)M_ * DF_ * 2));
    float*    z2     = (float*)   (ws + alloc((size_t)M_ * D_ * 4));

    // 1) conversions
    k_cvt_bf16<<<(M_ * D_ + 255) / 256, 256, 0, stream>>>(enc, x_bf, M_ * D_);
    k_cvt_bf16<<<(D_ * H_ * DV_ + 255) / 256, 256, 0, stream>>>(proj_w, pwb, D_ * H_ * DV_);
    k_cvt_bf16<<<(DF_ * D_ + 255) / 256, 256, 0, stream>>>(w1, w1b, DF_ * D_);
    k_cvt_bf16<<<(D_ * DF_ + 255) / 256, 256, 0, stream>>>(w2, w2b, D_ * DF_);
    k_cvt_tr_qkv<<<(H_ * D_ * DK_ + 255) / 256, 256, 0, stream>>>(w_qs, wqt, H_ * D_ * DK_);
    k_cvt_tr_qkv<<<(H_ * D_ * DK_ + 255) / 256, 256, 0, stream>>>(w_ks, wkt, H_ * D_ * DK_);
    k_cvt_tr_qkv<<<(H_ * D_ * DV_ + 255) / 256, 256, 0, stream>>>(w_vs, wvt, H_ * D_ * DV_);

    // 2) QKV projections: 3*8*512*2 = 24576 wave-tiles, 4 waves/block
    k_qkv<<<6144, 128, 0, stream>>>(x_bf, wqt, wkt, wvt, q_bf, k_bf, v_t);

    // 3) flash attention: 8*8*64 = 4096 wave-tiles
    k_attn<<<1024, 128, 0, stream>>>(q_bf, k_bf, v_t, mask, a_out);

    // 4) output proj + bias + residual -> z1
    k_gemm_ep<<<2048, 128, 0, stream>>>(a_out, pwb, proj_b, enc, z1, nullptr,
                                        M_, D_, H_ * DV_, 0);
    // 5) layer norm 1 -> out1 (f32 for residual, bf16 for FFN A)
    k_ln<<<2048, 128, 0, stream>>>(z1, ln1_a, ln1_b, out1f, out1b, M_);

    // 6) FFN1: relu(out1 @ w1^T + b1) -> hmid (bf16)
    k_gemm_ep<<<8192, 128, 0, stream>>>(out1b, w1b, b1, nullptr, nullptr, hmid,
                                        M_, DF_, D_, 1);
    // 7) FFN2: hmid @ w2^T + b2 + out1 -> z2
    k_gemm_ep<<<2048, 128, 0, stream>>>(hmid, w2b, b2, out1f, z2, nullptr,
                                        M_, D_, DF_, 0);
    // 8) layer norm 2 -> final output
    k_ln<<<2048, 128, 0, stream>>>(z2, ln2_a, ln2_b, (float*)d_out, nullptr, M_);
}
